// SimpleRewardModel_17076789968980
// MI455X (gfx1250) — matrix operand order repaired and checked by reading, not compiled, not run on hardware
//
#include <hip/hip_runtime.h>
#include <hip/hip_bf16.h>

typedef __attribute__((ext_vector_type(16))) _Float16 v16h;
typedef __attribute__((ext_vector_type(8)))  _Float16 v8h;
typedef __attribute__((ext_vector_type(4)))  _Float16 v4h;
typedef __attribute__((ext_vector_type(8)))  float    v8f;
typedef __attribute__((ext_vector_type(4)))  float    v4f;

#define B_  256
#define T_  512
#define E_  128
#define H_  64
#define G_  256   // 4*H

static __device__ __forceinline__ v8f wmma_f16(v16h a, v16h b, v8f c) {
    // D = A(16x32 f16) * B(32x16 f16) + C(16x16 f32)
    return __builtin_amdgcn_wmma_f32_16x16x32_f16(
        /*neg_a=*/false, a, /*neg_b=*/false, b,
        /*c_mod=*/(short)0, c, /*reuse_a=*/false, /*reuse_b=*/false);
}

static __device__ __forceinline__ float fast_sigmoid(float x) {
    return 1.0f / (1.0f + __expf(-x));
}
static __device__ __forceinline__ float fast_tanh(float x) {
    // 1 - 2/(e^{2x}+1): saturates correctly to +/-1 at extremes
    float e = __expf(2.0f * x);
    return 1.0f - 2.0f / (e + 1.0f);
}

// -------------------------------------------------------------------------
// Kernel 0: bulk f32 -> f16 conversion (emb table, W_ih, W_hh), 4 elems/thread
// -------------------------------------------------------------------------
__global__ __launch_bounds__(256) void k_cvt(
    const float* __restrict__ s, _Float16* __restrict__ d, int n4)
{
    const int i = blockIdx.x * 256 + threadIdx.x;
    if (i < n4) {
        const v4f x = *(const v4f*)(s + (size_t)i * 4);
        v4h y;
        #pragma unroll
        for (int e = 0; e < 4; ++e) y[e] = (_Float16)x[e];
        *(v4h*)(d + (size_t)i * 4) = y;
    }
}

// -------------------------------------------------------------------------
// Kernel 1: xg[t][btile][ntile] = (emb[ids] @ W_ih^T + b_ih + b_hh)
// stored as f16 in WMMA C-fragment order: [lane][r] (16B contiguous / lane)
// grid: (16 btiles, 64 t-chunks of 8), block 128 (wave w -> ntiles 4w..4w+3)
// All operands pre-converted to f16: inner loop is b128 loads + WMMA only.
// -------------------------------------------------------------------------
__global__ __launch_bounds__(128) void k_xg(
    const int* __restrict__ ids, const _Float16* __restrict__ embF,
    const _Float16* __restrict__ WihF, const float* __restrict__ bih,
    const float* __restrict__ bhh, _Float16* __restrict__ xg)
{
    const int lane  = threadIdx.x & 31;
    const int wave  = threadIdx.x >> 5;
    const int btile = blockIdx.x;
    const int t0    = blockIdx.y * 8;
    const int m     = lane & 15;
    const int hi    = lane >> 4;

    // B fragments of W_ih^T (K=128 -> 4 kfrags) for this wave's 4 n-tiles.
    // B(32x16) lane layout: col N=(l&15), element e -> K = kbase + 16*hi + e
    v16h Bf[4][4];
    float bias[4];
    #pragma unroll
    for (int j = 0; j < 4; ++j) {
        const int g = (wave * 4 + j) * 16 + m;          // output gate column
        #pragma unroll
        for (int kk = 0; kk < 4; ++kk) {
            const _Float16* src = WihF + g * E_ + kk * 32 + 16 * hi;
            const v8h lo = *(const v8h*)src;
            const v8h up = *(const v8h*)(src + 8);
            #pragma unroll
            for (int e = 0; e < 8; ++e) { Bf[j][kk][e] = lo[e]; Bf[j][kk][e + 8] = up[e]; }
        }
        bias[j] = bih[g] + bhh[g];
    }

    for (int tt = 0; tt < 8; ++tt) {
        const int t  = t0 + tt;
        const int id = ids[(btile * 16 + m) * T_ + t];
        const _Float16* row = embF + (size_t)id * E_;
        // A(16x32 f16) lane layout: row M=(l&15); element e -> K=kb+16*hi+e
        v16h A[4];
        #pragma unroll
        for (int kk = 0; kk < 4; ++kk) {
            const int kb = kk * 32 + 16 * hi;
            const v8h lo = *(const v8h*)(row + kb);
            const v8h up = *(const v8h*)(row + kb + 8);
            #pragma unroll
            for (int e = 0; e < 8; ++e) { A[kk][e] = lo[e]; A[kk][e + 8] = up[e]; }
        }
        #pragma unroll
        for (int j = 0; j < 4; ++j) {
            v8f acc = {};
            #pragma unroll
            for (int kk = 0; kk < 4; ++kk) acc = wmma_f16(A[kk], Bf[j][kk], acc);
            const int ng = wave * 4 + j;
            v8h out;
            #pragma unroll
            for (int r = 0; r < 8; ++r) out[r] = (_Float16)(acc[r] + bias[j]);
            const size_t base = ((((size_t)t * 16 + btile) * 16 + ng) << 8) + lane * 8;
            *(v8h*)(xg + base) = out;   // one global_store_b128 per tile
        }
    }
}

// -------------------------------------------------------------------------
// Kernel 2: sequential LSTM scan. grid 16 (btiles) x block 128 (4 waves).
// Wave w computes gate w (i/f/g/o), owns h/c columns [16w,16w+16).
// xg fragments double-buffered in registers (t+1 load issued during step t);
// t+2 lines pulled via global_prefetch.
// -------------------------------------------------------------------------
__global__ __launch_bounds__(128) void k_lstm(
    const _Float16* __restrict__ WhhF, const _Float16* __restrict__ xg,
    float* __restrict__ features)
{
    const int lane  = threadIdx.x & 31;
    const int wave  = threadIdx.x >> 5;      // == gate index
    const int btile = blockIdx.x;
    const int m     = lane & 15;
    const int hi    = lane >> 4;

    __shared__ float    gate_lds[4][4][256];  // [gate][16-col tile][lane*8+r]
    __shared__ _Float16 h_lds[16 * H_];       // h row-major [batch-row][H]

    // W_hh^T B fragments (K=64 -> 2 kfrags) for this wave's 4 gate n-tiles
    v16h Bh[4][2];
    #pragma unroll
    for (int j = 0; j < 4; ++j) {
        const int g = wave * 64 + j * 16 + m;
        #pragma unroll
        for (int kk = 0; kk < 2; ++kk) {
            const _Float16* src = WhhF + g * H_ + kk * 32 + 16 * hi;
            const v8h lo = *(const v8h*)src;
            const v8h up = *(const v8h*)(src + 8);
            #pragma unroll
            for (int e = 0; e < 8; ++e) { Bh[j][kk][e] = lo[e]; Bh[j][kk][e + 8] = up[e]; }
        }
    }

    v16h hA[2] = {};          // h == 0 at t=0
    float c[8], hsum[8];
    #pragma unroll
    for (int r = 0; r < 8; ++r) { c[r] = 0.0f; hsum[r] = 0.0f; }

    // fragment address helper: tiles (wave*4+j) are 512B apart, contiguous per wave
    const size_t lane_off = (size_t)lane * 8;
    #define FRAG_BASE(tt, jj) \
        (((((size_t)(tt) * 16 + btile) * 16 + wave * 4 + (jj)) << 8) + lane_off)

    v8h xh[4];
    #pragma unroll
    for (int j = 0; j < 4; ++j) xh[j] = *(const v8h*)(xg + FRAG_BASE(0, j));

    for (int t = 0; t < T_; ++t) {
        const int tn = (t + 1 < T_) ? t + 1 : t;
        const int tp = (t + 2 < T_) ? t + 2 : t;
        // prefetch t+2 fragments toward WGP (global_prefetch_b8)
        #pragma unroll
        for (int j = 0; j < 4; ++j)
            __builtin_prefetch(xg + FRAG_BASE(tp, j), 0, 0);

        // --- gate pre-activations: C = xg fragment, then 2 chained WMMAs;
        //     issue next step's fragment load as soon as this one is consumed
        float act[4][8];
        v8h xn[4];
        #pragma unroll
        for (int j = 0; j < 4; ++j) {
            v8f acc;
            #pragma unroll
            for (int r = 0; r < 8; ++r) acc[r] = (float)xh[j][r];
            xn[j] = *(const v8h*)(xg + FRAG_BASE(tn, j));   // pipelined load
            acc = wmma_f16(hA[0], Bh[j][0], acc);
            acc = wmma_f16(hA[1], Bh[j][1], acc);
            if (wave == 2) {                     // g-gate: tanh
                #pragma unroll
                for (int r = 0; r < 8; ++r) act[j][r] = fast_tanh(acc[r]);
            } else {                             // i/f/o: sigmoid
                #pragma unroll
                for (int r = 0; r < 8; ++r) act[j][r] = fast_sigmoid(acc[r]);
            }
        }
        #pragma unroll
        for (int j = 0; j < 4; ++j) {
            #pragma unroll
            for (int r = 0; r < 8; ++r) gate_lds[wave][j][lane * 8 + r] = act[j][r];
        }
        __syncthreads();

        // --- c/h update on this wave's 16-column slice of H (lane-aligned gather)
        #pragma unroll
        for (int r = 0; r < 8; ++r) {
            const float iv = gate_lds[0][wave][lane * 8 + r];
            const float fv = gate_lds[1][wave][lane * 8 + r];
            const float gv = gate_lds[2][wave][lane * 8 + r];
            const float ov = gate_lds[3][wave][lane * 8 + r];
            const float cn = fv * c[r] + iv * gv;
            c[r] = cn;
            const float hn = ov * fast_tanh(cn);
            hsum[r] += hn;
            h_lds[(r + 8 * hi) * H_ + wave * 16 + m] = (_Float16)hn;
        }
        __syncthreads();

        // --- rebuild h A-fragments from LDS (aligned ds_load_b128 x4)
        #pragma unroll
        for (int kk = 0; kk < 2; ++kk) {
            const _Float16* hr = h_lds + m * H_ + kk * 32 + 16 * hi;
            const v8h lo = *(const v8h*)hr;
            const v8h up = *(const v8h*)(hr + 8);
            #pragma unroll
            for (int e = 0; e < 8; ++e) { hA[kk][e] = lo[e]; hA[kk][e + 8] = up[e]; }
        }
        #pragma unroll
        for (int j = 0; j < 4; ++j) xh[j] = xn[j];
    }
    #undef FRAG_BASE

    const float inv = 1.0f / (float)T_;
    #pragma unroll
    for (int r = 0; r < 8; ++r) {
        const int brow = btile * 16 + r + 8 * hi;
        features[brow * H_ + wave * 16 + m] = hsum[r] * inv;
    }
}

// -------------------------------------------------------------------------
// Kernel 3: tiny MLP head. one block, one thread per batch row.
// -------------------------------------------------------------------------
__global__ __launch_bounds__(256) void k_head(
    const float* __restrict__ features, const float* __restrict__ W1,
    const float* __restrict__ b1, const float* __restrict__ W2,
    const float* __restrict__ b2, float* __restrict__ out)
{
    __shared__ float sW1[32 * 64];
    __shared__ float sW2[32];
    for (int i = threadIdx.x; i < 32 * 64; i += blockDim.x) sW1[i] = W1[i];
    if (threadIdx.x < 32) sW2[threadIdx.x] = W2[threadIdx.x];
    __syncthreads();

    const int b = blockIdx.x * blockDim.x + threadIdx.x;
    if (b >= B_) return;
    float f[64];
    #pragma unroll
    for (int k = 0; k < 64; ++k) f[k] = features[b * 64 + k];
    float acc = b2[0];
    for (int j = 0; j < 32; ++j) {
        float h = b1[j];
        #pragma unroll
        for (int k = 0; k < 64; ++k) h = fmaf(f[k], sW1[j * 64 + k], h);
        acc = fmaf(fmaxf(h, 0.0f), sW2[j], acc);
    }
    out[b] = acc;
}

extern "C" void kernel_launch(void* const* d_in, const int* in_sizes, int n_in,
                              void* d_out, int out_size, void* d_ws, size_t ws_size,
                              hipStream_t stream)
{
    const int*   ids = (const int*)  d_in[0];
    const float* emb = (const float*)d_in[1];
    const float* Wih = (const float*)d_in[2];
    const float* Whh = (const float*)d_in[3];
    const float* bih = (const float*)d_in[4];
    const float* bhh = (const float*)d_in[5];
    const float* W1  = (const float*)d_in[6];
    const float* b1  = (const float*)d_in[7];
    const float* W2  = (const float*)d_in[8];
    const float* b2  = (const float*)d_in[9];
    float* out = (float*)d_out;

    // workspace layout
    char* ws = (char*)d_ws;
    size_t off = 0;
    _Float16* xg = (_Float16*)(ws + off);               // 64 MB
    off += (size_t)T_ * 16 * 16 * 256 * sizeof(_Float16);
    float* features = (float*)(ws + off);               // 64 KB
    off += (size_t)B_ * H_ * sizeof(float);
    _Float16* embF = (_Float16*)(ws + off);             // 8 MB (32000*128)
    off += (size_t)32000 * E_ * sizeof(_Float16);
    _Float16* WihF = (_Float16*)(ws + off);             // 64 KB
    off += (size_t)G_ * E_ * sizeof(_Float16);
    _Float16* WhhF = (_Float16*)(ws + off);             // 32 KB
    off += (size_t)G_ * H_ * sizeof(_Float16);

    // one-time f32 -> f16 conversions
    {
        const int n4e = 32000 * E_ / 4;                 // 1,024,000
        k_cvt<<<(n4e + 255) / 256, 256, 0, stream>>>(emb, embF, n4e);
        const int n4i = G_ * E_ / 4;                    // 8,192
        k_cvt<<<(n4i + 255) / 256, 256, 0, stream>>>(Wih, WihF, n4i);
        const int n4h = G_ * H_ / 4;                    // 4,096
        k_cvt<<<(n4h + 255) / 256, 256, 0, stream>>>(Whh, WhhF, n4h);
    }

    dim3 g1(16, T_ / 8);
    k_xg  <<<g1, 128, 0, stream>>>(ids, embF, WihF, bih, bhh, xg);
    k_lstm<<<16, 128, 0, stream>>>(WhhF, xg, features);
    k_head<<<1, 256, 0, stream>>>(features, W1, b1, W2, b2, out);
}
